// VectorQuantizer_32444182954059
// MI455X (gfx1250) — matrix operand order repaired and checked by reading, not compiled, not run on hardware
//
#include <hip/hip_runtime.h>
#include <hip/hip_bf16.h>

// ---------------- problem constants ----------------
#define NUM_EMB   1024
#define EMB_DIM   256
#define N_ROWS    65536                 // 64 * 1024
#define QN        (N_ROWS * EMB_DIM)    // 16777216
#define QOFF      1
#define POFF      (QOFF + QN)           // 16777217
#define IOFF      (POFF + 1)            // 16777218

// ---------------- workspace layout (bytes) ----------------
#define WS_ENORM  0                     // 1024 f32
#define WS_HIST   4096                  // 1024 u32
#define WS_LOSS   8192                  // 512 f32
#define WS_BHI    16384                 // 1024*256 u16 (512 KB), WMMA-swizzled
#define WS_BLO    (16384 + 524288)      // 512 KB

typedef __attribute__((ext_vector_type(16))) __bf16 v16bf;
typedef __attribute__((ext_vector_type(8)))  float  v8f;

__device__ __forceinline__ unsigned short f2bf(float f) {
    unsigned u = __float_as_uint(f);
    u += 0x7FFFu + ((u >> 16) & 1u);          // round-to-nearest-even
    return (unsigned short)(u >> 16);
}

// ============================================================
// Kernel 1: per-code ||e||^2 + bf16 hi/lo split of emb, stored
// pre-swizzled in the WMMA B-matrix (32x16, bf16) lane layout:
//   element (kk, n) -> lane = n + 16*((kk>>3)&1), half h = (kk&7) + 8*(kk>>4)
// Fragment storage: frag = (c*8 + kc)*32 + lane, 16 halfs contiguous.
// ============================================================
__global__ void vq_prep(const float* __restrict__ emb,
                        float* __restrict__ enorm,
                        unsigned short* __restrict__ bhi,
                        unsigned short* __restrict__ blo) {
    const int k = blockIdx.x;       // code id 0..1023
    const int t = threadIdx.x;      // dim    0..255
    const float w = emb[k * EMB_DIM + t];

    // ||e||^2 reduction (deterministic tree)
    float s = w * w;
    #pragma unroll
    for (int m = 16; m; m >>= 1) s += __shfl_xor(s, m, 32);
    __shared__ float sh[8];
    if ((t & 31) == 0) sh[t >> 5] = s;
    __syncthreads();
    if (t == 0) {
        float tot = 0.f;
        #pragma unroll
        for (int i = 0; i < 8; ++i) tot += sh[i];
        enorm[k] = tot;
    }

    // bf16 hi/lo split
    const unsigned short h = f2bf(w);
    const float hf = __uint_as_float(((unsigned)h) << 16);
    const unsigned short l = f2bf(w - hf);

    // swizzle into B fragment layout
    const int c  = k >> 4;          // code chunk (16 codes)
    const int n  = k & 15;          // column within chunk
    const int kc = t >> 5;          // 32-wide contraction chunk
    const int kk = t & 31;          // element within chunk
    const int lane = n + (((kk >> 3) & 1) << 4);
    const int hh   = (kk & 7) + ((kk >> 4) << 3);
    const int off  = (((c * 8 + kc) * 32 + lane) << 4) + hh;
    bhi[off] = h;
    blo[off] = l;
}

// ============================================================
// Kernel 2: zero the code histogram (must happen every call)
// ============================================================
__global__ void vq_hist_zero(unsigned* __restrict__ hist) {
    hist[blockIdx.x * 256 + threadIdx.x] = 0u;
}

// ============================================================
// Kernel 3: main — WMMA distance GEMM + argmin + gather + loss
// 512 blocks x 256 threads (8 waves); wave owns 16 rows.
// ============================================================
__global__ void __launch_bounds__(256)
vq_main(const float* __restrict__ x,
        const float* __restrict__ emb,
        const float* __restrict__ enorm,
        const unsigned short* __restrict__ bhiraw,
        const unsigned short* __restrict__ bloraw,
        unsigned* __restrict__ hist,
        float* __restrict__ lossp,
        float* __restrict__ out) {
    const int tid  = threadIdx.x;
    const int wave = tid >> 5;
    const int lane = tid & 31;
    const int lg   = lane & 15;     // lane within 16-group
    const int g    = lane >> 4;     // group 0/1
    const int rowbase = blockIdx.x * 128 + wave * 16;

    // ---- build A fragments (16-bit A 16x32 layout):
    // lane = M + 16*((K>>3)&1); halfs h=0..7 <- K run {base..base+7},
    // h=8..15 <- K run {base+16..base+23}, base = kc*32 + 8*g
    const int arow = rowbase + lg;                 // A row M = lg
    const float4* xr = (const float4*)(x + (size_t)arow * EMB_DIM);

    v16bf Ahi[8], Alo[8];
    #pragma unroll
    for (int kc = 0; kc < 8; ++kc) {
        const int b0 = kc * 8 + g * 2;             // float4 index
        float f[16];
        float4 p;
        p = xr[b0];     f[0]=p.x;  f[1]=p.y;  f[2]=p.z;  f[3]=p.w;
        p = xr[b0 + 1]; f[4]=p.x;  f[5]=p.y;  f[6]=p.z;  f[7]=p.w;
        p = xr[b0 + 4]; f[8]=p.x;  f[9]=p.y;  f[10]=p.z; f[11]=p.w;
        p = xr[b0 + 5]; f[12]=p.x; f[13]=p.y; f[14]=p.z; f[15]=p.w;
        union { v16bf v; unsigned short s[16]; } uh, ul;
        #pragma unroll
        for (int j = 0; j < 16; ++j) {
            const unsigned short hb = f2bf(f[j]);
            uh.s[j] = hb;
            ul.s[j] = f2bf(f[j] - __uint_as_float(((unsigned)hb) << 16));
        }
        Ahi[kc] = uh.v;
        Alo[kc] = ul.v;
    }

    const v16bf* Bhi = (const v16bf*)bhiraw;
    const v16bf* Blo = (const v16bf*)bloraw;

    float minv[8];
    int   mini[8];
    #pragma unroll
    for (int r = 0; r < 8; ++r) { minv[r] = 3.4e38f; mini[r] = 0; }

    // ---- 64 code chunks; 8 k-chunks x 3 WMMAs each (bf16x3 split)
    for (int c = 0; c < 64; ++c) {
        v8f acc = {};
        #pragma unroll
        for (int kc = 0; kc < 8; ++kc) {
            const int frag = (c * 8 + kc) * 32 + lane;
            const v16bf bh = Bhi[frag];
            const v16bf bl = Blo[frag];
            acc = __builtin_amdgcn_wmma_f32_16x16x32_bf16(
                      false, Ahi[kc], false, bh, (short)0, acc, false, false);
            acc = __builtin_amdgcn_wmma_f32_16x16x32_bf16(
                      false, Ahi[kc], false, bl, (short)0, acc, false, false);
            acc = __builtin_amdgcn_wmma_f32_16x16x32_bf16(
                      false, Alo[kc], false, bh, (short)0, acc, false, false);
        }
        // distance (minus constant ||x||^2): ||e||^2 - 2 x.e
        const int   code = c * 16 + lg;       // C layout: N = lane&15
        const float en   = enorm[code];
        #pragma unroll
        for (int r = 0; r < 8; ++r) {
            const float d = fmaf(-2.0f, acc[r], en);
            if (d < minv[r]) { minv[r] = d; mini[r] = code; }
        }
    }

    // ---- cross-lane argmin within each 16-lane group
    // (C layout: lanes 0-15 carry rows M=r, lanes 16-31 rows M=8+r)
    #pragma unroll
    for (int r = 0; r < 8; ++r) {
        #pragma unroll
        for (int m = 8; m; m >>= 1) {
            const float ov = __shfl_xor(minv[r], m, 16);
            const int   oi = __shfl_xor(mini[r], m, 16);
            if (ov < minv[r] || (ov == minv[r] && oi < mini[r])) {
                minv[r] = ov; mini[r] = oi;
            }
        }
    }

    // ---- indices + histogram (integer atomics => deterministic)
    if (lg == 0) {
        #pragma unroll
        for (int r = 0; r < 8; ++r) {
            const int row = rowbase + g * 8 + r;
            out[IOFF + row] = (float)mini[r];
            atomicAdd(&hist[mini[r]], 1u);
        }
    }

    // ---- gather quantized rows + MSE partial
    // group g handles rows {8g..8g+7}; 16 lanes stride the 256 dims
    float ls = 0.f;
    #pragma unroll
    for (int r = 0; r < 8; ++r) {
        const int row  = rowbase + g * 8 + r;
        const int code = mini[r];                       // uniform in group
        const float* e  = emb + (size_t)code * EMB_DIM;
        const float* xi = x   + (size_t)row  * EMB_DIM;
        float*       q  = out + QOFF + (size_t)row * EMB_DIM;
        #pragma unroll
        for (int j = 0; j < 16; ++j) {
            const int ecol = j * 16 + lg;               // coalesced stores
            const float qv = e[ecol];
            const float xv = xi[ecol];
            q[ecol] = qv;                                // == quantized_st
            const float d = qv - xv;
            ls += d * d;
        }
    }

    // ---- deterministic loss reduction: shfl tree + fixed-order LDS sum
    #pragma unroll
    for (int m = 16; m; m >>= 1) ls += __shfl_xor(ls, m, 32);
    __shared__ float shl[8];
    if (lane == 0) shl[wave] = ls;
    __syncthreads();
    if (tid == 0) {
        float t = 0.f;
        #pragma unroll
        for (int i = 0; i < 8; ++i) t += shl[i];
        lossp[blockIdx.x] = t;
    }
}

// ============================================================
// Kernel 4: final scalars (loss, perplexity), fixed-order trees
// ============================================================
__global__ void vq_final(const unsigned* __restrict__ hist,
                         const float* __restrict__ lossp,
                         float* __restrict__ out) {
    const int t = threadIdx.x;      // 1024 threads
    __shared__ float sd[1024];

    // loss = 1.25 * sum / (N*D)
    sd[t] = (t < 512) ? lossp[t] : 0.f;
    __syncthreads();
    for (int s = 512; s; s >>= 1) {
        if (t < s) sd[t] += sd[t + s];
        __syncthreads();
    }
    if (t == 0) out[0] = 1.25f * sd[0] / (float)QN;
    __syncthreads();

    // perplexity = exp(-sum p*log(p+1e-10)), p = count / N
    const float p = (float)hist[t] * (1.0f / (float)N_ROWS);
    sd[t] = p * logf(p + 1e-10f);
    __syncthreads();
    for (int s = 512; s; s >>= 1) {
        if (t < s) sd[t] += sd[t + s];
        __syncthreads();
    }
    if (t == 0) out[POFF] = expf(-sd[0]);
}

// ============================================================
extern "C" void kernel_launch(void* const* d_in, const int* in_sizes, int n_in,
                              void* d_out, int out_size, void* d_ws, size_t ws_size,
                              hipStream_t stream) {
    const float* x   = (const float*)d_in[0];   // [64,1024,256] f32
    const float* emb = (const float*)d_in[1];   // [1024,256]    f32
    float* out = (float*)d_out;
    char*  ws  = (char*)d_ws;

    float*          enorm = (float*)(ws + WS_ENORM);
    unsigned*       hist  = (unsigned*)(ws + WS_HIST);
    float*          lossp = (float*)(ws + WS_LOSS);
    unsigned short* bhi   = (unsigned short*)(ws + WS_BHI);
    unsigned short* blo   = (unsigned short*)(ws + WS_BLO);

    vq_prep<<<NUM_EMB, EMB_DIM, 0, stream>>>(emb, enorm, bhi, blo);
    vq_hist_zero<<<NUM_EMB / 256, 256, 0, stream>>>(hist);
    vq_main<<<N_ROWS / 128, 256, 0, stream>>>(x, emb, enorm, bhi, blo,
                                              hist, lossp, out);
    vq_final<<<1, 1024, 0, stream>>>(hist, lossp, out);
}
